// HailNet_86775519248758
// MI455X (gfx1250) — matrix operand order repaired and checked by reading, not compiled, not run on hardware
//
#include <hip/hip_runtime.h>
#include <hip/hip_bf16.h>
#include <stdint.h>

#define T_STEPS 12
#define BATCH   1024
#define NGRID   4096
#define DFEAT   256
#define HDIM    256
#define TBROWS  (T_STEPS * BATCH)   // 12288
#define G3H     (3 * HDIM)          // 768

typedef __bf16 bf16_t;
typedef __attribute__((ext_vector_type(16))) __bf16 bf16x16;
typedef __attribute__((ext_vector_type(8)))  float  f32x8;

__device__ __forceinline__ float sigmoid_f(float x) {
  return 1.0f / (1.0f + __expf(-x));
}

// Async DMA: global -> LDS, 16 bytes per lane, tracked by ASYNCcnt.
// LDS address = low 32 bits of the generic pointer (flat-aperture rule).
__device__ __forceinline__ void async_copy_b128(void* lds_dst, const void* gsrc) {
  uint32_t l = (uint32_t)(uintptr_t)lds_dst;
  asm volatile("global_load_async_to_lds_b128 %0, %1, off"
               :: "v"(l), "v"(gsrc) : "memory");
}
__device__ __forceinline__ void wait_async0() {
  asm volatile("s_wait_asynccnt 0x0" ::: "memory");
}

// ---------------------------------------------------------------------------
// fp32 -> bf16 convert
// ---------------------------------------------------------------------------
__global__ __launch_bounds__(256) void k_f2bf(const float* __restrict__ s,
                                              bf16_t* __restrict__ d, int n) {
  int i = blockIdx.x * 256 + threadIdx.x;
  if (i < n) d[i] = (bf16_t)s[i];
}

// ---------------------------------------------------------------------------
// W_eff[d][n] = sum_m W_gnn[d][m] * A[m][n]; A nonzeros only at
// m = n + {0, +-1, +-63, +-64, +-65}. Reads true A values (duplicate edge
// weights exact). Output bf16.
// ---------------------------------------------------------------------------
__global__ __launch_bounds__(256) void k_weff(const float* __restrict__ A,
                                              const float* __restrict__ Wg,
                                              bf16_t* __restrict__ Weff) {
  int idx = blockIdx.x * 256 + threadIdx.x;   // 256*4096 threads
  int d = idx >> 12;
  int n = idx & (NGRID - 1);
  const int offs[9] = {0, 1, -1, 63, -63, 64, -64, 65, -65};
  float acc = 0.0f;
#pragma unroll
  for (int o = 0; o < 9; ++o) {
    int m = n + offs[o];
    if (m >= 0 && m < NGRID)
      acc += A[(size_t)m * NGRID + n] * Wg[(size_t)d * NGRID + m];
  }
  Weff[(size_t)d * NGRID + n] = (bf16_t)acc;
}

// ---------------------------------------------------------------------------
// hidden-state init: fp32 + bf16 copies of h0[0], h0[1]
// ---------------------------------------------------------------------------
__global__ __launch_bounds__(256) void k_inith(const float* __restrict__ h0,
                                               float* __restrict__ hc0,
                                               float* __restrict__ hc1,
                                               bf16_t* __restrict__ hb0,
                                               bf16_t* __restrict__ hb1) {
  int i = blockIdx.x * 256 + threadIdx.x;     // BATCH*HDIM
  if (i < BATCH * HDIM) {
    float a = h0[i];
    float b = h0[BATCH * HDIM + i];
    hc0[i] = a; hc1[i] = b;
    hb0[i] = (bf16_t)a; hb1[i] = (bf16_t)b;
  }
}

// ---------------------------------------------------------------------------
// Fused GNN: t4 = sigmoid(sigmoid(x @ W_eff^T + b_gnn) @ W_lin^T + b_lin)
// Block: 64 rows x 256 cols, 8 waves; wave w owns cols [32w, 32w+32).
// Software-pipelined: double-buffered LDS; W tiles via async DMA issued
// before the WMMA phase; X next-tile in registers during compute.
// T2s (phase-2 input) aliases the phase-1 staging buffers via a union.
// ---------------------------------------------------------------------------
struct __align__(16) SmemGnn {
  union {
    struct {
      bf16_t Xs[2][64][40];    // 2 x 64x32 bf16 x tiles (padded)
      bf16_t Ws[2][256][40];   // 2 x 256x32 bf16 W_eff tiles
    } p;
    bf16_t T2s[64][264];       // 64x256 bf16 sigmoid(GEMM1)
  };
};

__global__ __launch_bounds__(256)
void k_gnn(const float* __restrict__ x, const bf16_t* __restrict__ Weff,
           const float* __restrict__ bgnn, const bf16_t* __restrict__ Wlin,
           const float* __restrict__ blin, bf16_t* __restrict__ t4) {
  __shared__ SmemGnn sm;

  const int tid = threadIdx.x;
  const int wave = tid >> 5;
  const int lane = tid & 31;
  const int lh = lane >> 4;         // lane half
  const int ll = lane & 15;
  const int rowBlk = blockIdx.x * 64;
  const int colBase = wave * 32;
  const int KTILES = NGRID / 32;    // 128

  // staging helpers ---------------------------------------------------------
  float4 xr[2];
  auto issueW = [&](int kt, int buf) {       // 4 async b128 per thread
    int kk = kt * 32;
#pragma unroll
    for (int it = 0; it < 4; ++it) {
      int id = tid + it * 256;               // 0..1023
      int n = id >> 2;
      int c = (id & 3) << 3;
      async_copy_b128(&sm.p.Ws[buf][n][c], Weff + (size_t)n * NGRID + kk + c);
    }
  };
  auto loadX = [&](int kt) {                 // 2 float4 per thread
    int kk = kt * 32;
#pragma unroll
    for (int it = 0; it < 2; ++it) {
      int id = tid + it * 256;               // 0..511
      int r = id >> 3;
      int c = (id & 7) << 2;
      xr[it] = *reinterpret_cast<const float4*>(
          x + (size_t)(rowBlk + r) * NGRID + kk + c);
    }
  };
  auto storeX = [&](int buf) {               // cvt fp32->bf16 into LDS
#pragma unroll
    for (int it = 0; it < 2; ++it) {
      int id = tid + it * 256;
      int r = id >> 3;
      int c = (id & 7) << 2;
      sm.p.Xs[buf][r][c + 0] = (bf16_t)xr[it].x;
      sm.p.Xs[buf][r][c + 1] = (bf16_t)xr[it].y;
      sm.p.Xs[buf][r][c + 2] = (bf16_t)xr[it].z;
      sm.p.Xs[buf][r][c + 3] = (bf16_t)xr[it].w;
    }
  };

  // prologue: stage tile 0
  issueW(0, 0);
  loadX(0);
  storeX(0);
  wait_async0();
  __syncthreads();

  f32x8 acc[4][2] = {};
  for (int kt = 0; kt < KTILES; ++kt) {
    const int buf = kt & 1;
    const bool more = (kt + 1) < KTILES;
    if (more) {                      // overlap next-tile DMA with compute
      issueW(kt + 1, buf ^ 1);
      loadX(kt + 1);
      __builtin_prefetch(x + (size_t)(rowBlk + (tid & 63)) * NGRID +
                         (kt + 2) * 32, 0, 1);
    }
    // WMMA phase on current buffers
    bf16x16 bfrag[2];
#pragma unroll
    for (int nt = 0; nt < 2; ++nt) {
      const bf16_t* q = &sm.p.Ws[buf][colBase + nt * 16 + ll][lh << 4];
#pragma unroll
      for (int i = 0; i < 16; ++i) bfrag[nt][i] = q[i];
    }
#pragma unroll
    for (int mt = 0; mt < 4; ++mt) {
      bf16x16 a;
      const bf16_t* p0 = &sm.p.Xs[buf][mt * 16 + ll][lh << 4];
#pragma unroll
      for (int i = 0; i < 16; ++i) a[i] = p0[i];
#pragma unroll
      for (int nt = 0; nt < 2; ++nt)
        acc[mt][nt] = __builtin_amdgcn_wmma_f32_16x16x32_bf16(
            false, a, false, bfrag[nt], (short)0, acc[mt][nt], false, false);
    }
    if (more) storeX(buf ^ 1);
    wait_async0();                   // own DMA for tile kt+1 has landed
    __syncthreads();                 // everyone's stores visible
  }

  // epilogue 1: sigmoid(+b_gnn) -> T2s (aliases staging buffers; all reads
  // of those buffers finished before the loop's final barrier)
#pragma unroll
  for (int mt = 0; mt < 4; ++mt)
#pragma unroll
    for (int nt = 0; nt < 2; ++nt) {
      int n = colBase + nt * 16 + ll;
      float bb = bgnn[n];
#pragma unroll
      for (int r = 0; r < 8; ++r) {
        int m = mt * 16 + r + (lh << 3);
        sm.T2s[m][n] = (bf16_t)sigmoid_f(acc[mt][nt][r] + bb);
      }
    }
  __syncthreads();

  // GEMM2: t4 = sigmoid(T2 @ W_lin^T + b_lin), K = 256
  f32x8 acc2[4][2] = {};
  for (int kk = 0; kk < DFEAT; kk += 32) {
    bf16x16 bfrag[2];
#pragma unroll
    for (int nt = 0; nt < 2; ++nt) {
      const bf16_t* q =
          Wlin + (size_t)(colBase + nt * 16 + ll) * DFEAT + kk + (lh << 4);
#pragma unroll
      for (int i = 0; i < 16; ++i) bfrag[nt][i] = q[i];
    }
#pragma unroll
    for (int mt = 0; mt < 4; ++mt) {
      bf16x16 a;
      const bf16_t* p0 = &sm.T2s[mt * 16 + ll][kk + (lh << 4)];
#pragma unroll
      for (int i = 0; i < 16; ++i) a[i] = p0[i];
#pragma unroll
      for (int nt = 0; nt < 2; ++nt)
        acc2[mt][nt] = __builtin_amdgcn_wmma_f32_16x16x32_bf16(
            false, a, false, bfrag[nt], (short)0, acc2[mt][nt], false, false);
    }
  }
#pragma unroll
  for (int mt = 0; mt < 4; ++mt)
#pragma unroll
    for (int nt = 0; nt < 2; ++nt) {
      int n = colBase + nt * 16 + ll;
      float bb = blin[n];
#pragma unroll
      for (int r = 0; r < 8; ++r) {
        int m = mt * 16 + r + (lh << 3);
        t4[(size_t)(rowBlk + m) * DFEAT + n] =
            (bf16_t)sigmoid_f(acc2[mt][nt][r] + bb);
      }
    }
}

// ---------------------------------------------------------------------------
// C[M][Nout] (fp32) = A_bf16[M][256] @ W_bf16[Nout][256]^T + bias
// Block tile 64x64; 8 waves as 4(M) x 2(N); K fixed = 256. A tile staged
// with async-to-LDS DMA.
// ---------------------------------------------------------------------------
__global__ __launch_bounds__(256)
void k_gemm_bias(const bf16_t* __restrict__ Am, const bf16_t* __restrict__ W,
                 const float* __restrict__ bias, float* __restrict__ C,
                 int Nout) {
  __shared__ __align__(16) bf16_t As[64][40];
  const int tid = threadIdx.x;
  const int wave = tid >> 5, lane = tid & 31;
  const int lh = lane >> 4, ll = lane & 15;
  const int wm = wave >> 1, wn = wave & 1;
  const int rowBlk = blockIdx.x * 64;
  const int colBlk = blockIdx.y * 64;

  f32x8 acc[2] = {};
  for (int kk = 0; kk < 256; kk += 32) {
    int r = tid >> 2, c = (tid & 3) << 3;       // 64x32 tile, 1 b128/thread
    async_copy_b128(&As[r][c], Am + (size_t)(rowBlk + r) * 256 + kk + c);
    wait_async0();
    __syncthreads();
    bf16x16 a;
    const bf16_t* p0 = &As[wm * 16 + ll][lh << 4];
#pragma unroll
    for (int i = 0; i < 16; ++i) a[i] = p0[i];
#pragma unroll
    for (int nt = 0; nt < 2; ++nt) {
      int n = colBlk + wn * 32 + nt * 16 + ll;
      const bf16_t* q = W + (size_t)n * 256 + kk + (lh << 4);
      bf16x16 b;
#pragma unroll
      for (int i = 0; i < 16; ++i) b[i] = q[i];
      acc[nt] = __builtin_amdgcn_wmma_f32_16x16x32_bf16(
          false, a, false, b, (short)0, acc[nt], false, false);
    }
    __syncthreads();
  }
#pragma unroll
  for (int nt = 0; nt < 2; ++nt) {
    int n = colBlk + wn * 32 + nt * 16 + ll;
    float bb = bias[n];
#pragma unroll
    for (int r = 0; r < 8; ++r) {
      int m = wm * 16 + r + (lh << 3);
      C[(size_t)(rowBlk + m) * Nout + n] = acc[nt][r] + bb;
    }
  }
}

// ---------------------------------------------------------------------------
// One GRU step: hW = h_prev @ Whh^T (+bhh), gates, h_new.
// Block: 16 batch rows; 8 waves, wave w owns cols [96w, 96w+96) of hW.
// ---------------------------------------------------------------------------
__global__ __launch_bounds__(256)
void k_gru(const float* __restrict__ xW,          // [BATCH][768], incl. bih
           const bf16_t* __restrict__ hprev_bf,   // [BATCH][256]
           const float* __restrict__ hprev_f,     // [BATCH][256]
           const bf16_t* __restrict__ Whh,        // [768][256] bf16
           const float* __restrict__ bhh,         // [768]
           float* __restrict__ hnew_f, bf16_t* __restrict__ hnew_bf) {
  __shared__ __align__(16) bf16_t Hs[16][40];
  __shared__ float hWs[16][776];
  const int tid = threadIdx.x;
  const int wave = tid >> 5, lane = tid & 31;
  const int lh = lane >> 4, ll = lane & 15;
  const int rowBlk = blockIdx.x * 16;

  f32x8 acc[6] = {};
  for (int kk = 0; kk < 256; kk += 32) {
    if (tid < 64) {
      int r = tid >> 2, c = (tid & 3) << 3;
      async_copy_b128(&Hs[r][c], hprev_bf + (size_t)(rowBlk + r) * 256 + kk + c);
    }
    wait_async0();
    __syncthreads();
    bf16x16 a;
    const bf16_t* p0 = &Hs[ll][lh << 4];
#pragma unroll
    for (int i = 0; i < 16; ++i) a[i] = p0[i];
#pragma unroll
    for (int nt = 0; nt < 6; ++nt) {
      int n = wave * 96 + nt * 16 + ll;
      const bf16_t* q = Whh + (size_t)n * 256 + kk + (lh << 4);
      bf16x16 b;
#pragma unroll
      for (int i = 0; i < 16; ++i) b[i] = q[i];
      acc[nt] = __builtin_amdgcn_wmma_f32_16x16x32_bf16(
          false, a, false, b, (short)0, acc[nt], false, false);
    }
    __syncthreads();
  }
#pragma unroll
  for (int nt = 0; nt < 6; ++nt) {
    int n = wave * 96 + nt * 16 + ll;
#pragma unroll
    for (int r = 0; r < 8; ++r) hWs[r + (lh << 3)][n] = acc[nt][r];
  }
  __syncthreads();

  // gates: 16 rows x 256 -> 16 elements per thread
#pragma unroll
  for (int i = 0; i < 16; ++i) {
    int idx = tid + i * 256;
    int r = idx >> 8, j = idx & 255;
    int b = rowBlk + r;
    float rh = hWs[r][j] + bhh[j];
    float zh = hWs[r][j + 256] + bhh[j + 256];
    float nh = hWs[r][j + 512] + bhh[j + 512];
    const float* xwr = xW + (size_t)b * G3H;
    float rr = sigmoid_f(xwr[j] + rh);
    float zz = sigmoid_f(xwr[j + 256] + zh);
    float nn = tanhf(xwr[j + 512] + rr * nh);
    float hp = hprev_f[(size_t)b * 256 + j];
    float hn = (1.0f - zz) * nn + zz * hp;
    hnew_f[(size_t)b * 256 + j] = hn;
    hnew_bf[(size_t)b * 256 + j] = (bf16_t)hn;
  }
}

// ---------------------------------------------------------------------------
// Head MLP: 256 -> 16 -> 16 -> 1, sigmoid after each. One thread per batch.
// ---------------------------------------------------------------------------
__global__ __launch_bounds__(256)
void k_head(const float* __restrict__ hL, const float* __restrict__ Wf0,
            const float* __restrict__ b0, const float* __restrict__ Wf1,
            const float* __restrict__ b1, const float* __restrict__ Wf2,
            const float* __restrict__ b2, float* __restrict__ out) {
  int b = blockIdx.x * 256 + threadIdx.x;
  if (b >= BATCH) return;
  float v0[16];
#pragma unroll
  for (int i = 0; i < 16; ++i) v0[i] = b0[i];
  const float* hr = hL + (size_t)b * 256;
  for (int k = 0; k < 256; k += 4) {
    float4 hv = *reinterpret_cast<const float4*>(&hr[k]);
#pragma unroll
    for (int i = 0; i < 16; ++i) {
      const float* w = Wf0 + i * 256 + k;
      v0[i] += hv.x * w[0] + hv.y * w[1] + hv.z * w[2] + hv.w * w[3];
    }
  }
#pragma unroll
  for (int i = 0; i < 16; ++i) v0[i] = sigmoid_f(v0[i]);
  float v1[16];
#pragma unroll
  for (int i = 0; i < 16; ++i) {
    float s = b1[i];
#pragma unroll
    for (int j = 0; j < 16; ++j) s += Wf1[i * 16 + j] * v0[j];
    v1[i] = sigmoid_f(s);
  }
  float o = b2[0];
#pragma unroll
  for (int j = 0; j < 16; ++j) o += Wf2[j] * v1[j];
  out[b] = sigmoid_f(o);
}

// ---------------------------------------------------------------------------
extern "C" void kernel_launch(void* const* d_in, const int* in_sizes, int n_in,
                              void* d_out, int out_size, void* d_ws, size_t ws_size,
                              hipStream_t stream) {
  (void)in_sizes; (void)n_in; (void)out_size; (void)ws_size;
  const float* x    = (const float*)d_in[0];
  const float* h0   = (const float*)d_in[1];
  const float* A    = (const float*)d_in[2];
  const float* Wgnn = (const float*)d_in[3];
  const float* bgnn = (const float*)d_in[4];
  const float* Wlin = (const float*)d_in[5];
  const float* blin = (const float*)d_in[6];
  const float* Wih0 = (const float*)d_in[7];
  const float* Whh0 = (const float*)d_in[8];
  const float* bih0 = (const float*)d_in[9];
  const float* bhh0 = (const float*)d_in[10];
  const float* Wih1 = (const float*)d_in[11];
  const float* Whh1 = (const float*)d_in[12];
  const float* bih1 = (const float*)d_in[13];
  const float* bhh1 = (const float*)d_in[14];
  const float* Wf0  = (const float*)d_in[15];
  const float* bf0  = (const float*)d_in[16];
  const float* Wf1  = (const float*)d_in[17];
  const float* bf1  = (const float*)d_in[18];
  const float* Wf2  = (const float*)d_in[19];
  const float* bf2  = (const float*)d_in[20];
  float* out = (float*)d_out;

  uint8_t* wsb = (uint8_t*)d_ws;
  size_t off = 0;
  auto wsAlloc = [&](size_t bytes) -> void* {
    void* p = wsb + off;
    off += (bytes + 255) & ~(size_t)255;
    return p;
  };
  bf16_t* WeffB = (bf16_t*)wsAlloc((size_t)DFEAT * NGRID * 2);
  bf16_t* WlinB = (bf16_t*)wsAlloc((size_t)DFEAT * DFEAT * 2);
  bf16_t* Wih0B = (bf16_t*)wsAlloc((size_t)G3H * HDIM * 2);
  bf16_t* Whh0B = (bf16_t*)wsAlloc((size_t)G3H * HDIM * 2);
  bf16_t* Wih1B = (bf16_t*)wsAlloc((size_t)G3H * HDIM * 2);
  bf16_t* Whh1B = (bf16_t*)wsAlloc((size_t)G3H * HDIM * 2);
  bf16_t* t4B   = (bf16_t*)wsAlloc((size_t)TBROWS * DFEAT * 2);
  bf16_t* hs0B  = (bf16_t*)wsAlloc((size_t)TBROWS * HDIM * 2);
  bf16_t* h0b0  = (bf16_t*)wsAlloc((size_t)BATCH * HDIM * 2);
  bf16_t* h0b1  = (bf16_t*)wsAlloc((size_t)BATCH * HDIM * 2);
  bf16_t* hb1   = (bf16_t*)wsAlloc((size_t)BATCH * HDIM * 2);
  float*  hc0   = (float*)wsAlloc((size_t)BATCH * HDIM * 4);
  float*  hc1   = (float*)wsAlloc((size_t)BATCH * HDIM * 4);
  float*  xwbuf = (float*)wsAlloc((size_t)TBROWS * G3H * 4);

  auto cvt = [&](const float* s, bf16_t* d, int n) {
    k_f2bf<<<(n + 255) / 256, 256, 0, stream>>>(s, d, n);
  };
  cvt(Wlin, WlinB, DFEAT * DFEAT);
  cvt(Wih0, Wih0B, G3H * HDIM);
  cvt(Whh0, Whh0B, G3H * HDIM);
  cvt(Wih1, Wih1B, G3H * HDIM);
  cvt(Whh1, Whh1B, G3H * HDIM);

  k_weff<<<(DFEAT * NGRID) / 256, 256, 0, stream>>>(A, Wgnn, WeffB);
  k_inith<<<(BATCH * HDIM) / 256, 256, 0, stream>>>(h0, hc0, hc1, h0b0, h0b1);

  // fused GNN + HailNet lin1 -> t4 bf16
  k_gnn<<<TBROWS / 64, 256, 0, stream>>>(x, WeffB, bgnn, WlinB, blin, t4B);

  // layer-0 input projection (all timesteps batched)
  k_gemm_bias<<<dim3(TBROWS / 64, G3H / 64), 256, 0, stream>>>(
      t4B, Wih0B, bih0, xwbuf, G3H);

  for (int t = 0; t < T_STEPS; ++t) {
    const float* xw = xwbuf + (size_t)t * BATCH * G3H;
    const bf16_t* hpb = (t == 0) ? h0b0 : hs0B + (size_t)(t - 1) * BATCH * HDIM;
    k_gru<<<BATCH / 16, 256, 0, stream>>>(xw, hpb, hc0, Whh0B, bhh0, hc0,
                                          hs0B + (size_t)t * BATCH * HDIM);
  }

  // layer-1 input projection (all timesteps batched)
  k_gemm_bias<<<dim3(TBROWS / 64, G3H / 64), 256, 0, stream>>>(
      hs0B, Wih1B, bih1, xwbuf, G3H);

  for (int t = 0; t < T_STEPS; ++t) {
    const float* xw = xwbuf + (size_t)t * BATCH * G3H;
    const bf16_t* hpb = (t == 0) ? h0b1 : hb1;
    k_gru<<<BATCH / 16, 256, 0, stream>>>(xw, hpb, hc1, Whh1B, bhh1, hc1, hb1);
  }

  k_head<<<BATCH / 256, 256, 0, stream>>>(hc1, Wf0, bf0, Wf1, bf1, Wf2, bf2, out);
}
